// MultiheadAttention_6296422055961
// MI455X (gfx1250) — matrix operand order repaired and checked by reading, not compiled, not run on hardware
//
#include <hip/hip_runtime.h>

typedef __attribute__((ext_vector_type(16))) __bf16 v16bf;
typedef __attribute__((ext_vector_type(4)))  __bf16 v4bf;
typedef __attribute__((ext_vector_type(8)))  float  v8f;
typedef __attribute__((ext_vector_type(4)))  float  v4f;

#define IN_DIM   1024
#define MROWS    8192                              // B*S = 8*1024
#define YBYTES   (size_t)(MROWS * IN_DIM * 2)      // one bf16 intermediate matrix
#define WBYTES   (size_t)(IN_DIM * IN_DIM * 2)     // one bf16 weight matrix

// ---------------------------------------------------------------------------
// One-shot fp32 -> bf16 weight conversion, vectorized (float4 NT load ->
// paired v_cvt_pk_bf16_f32 -> 8B store). fp32 weights are never re-read:
// NT loads keep them out of L2 so the bf16 intermediates stay resident.
// ---------------------------------------------------------------------------
__global__ __launch_bounds__(256)
void cvt_f32_bf16(const float* __restrict__ src, __bf16* __restrict__ dst, int n4)
{
    const v4f* s4 = (const v4f*)src;
    v4bf*      d4 = (v4bf*)dst;
    for (int i = blockIdx.x * 256 + threadIdx.x; i < n4; i += gridDim.x * 256) {
        v4f f = __builtin_nontemporal_load(s4 + i);
        d4[i] = __builtin_convertvector(f, v4bf);
    }
}

// ---------------------------------------------------------------------------
// bf16 WMMA GEMM:  C[M,N] = act( A[M,K] @ W[N,K]^T + bias[N] )
//   A fp32 (inputs; NT b128 loads + paired cvt_pk in-register) or bf16
//   (intermediate, L2-resident, RT loads). W pre-converted bf16 (L2-resident).
//   Workgroup tile 128(M) x 256(N); 8 waves as 2(M) x 4(N); wave tile 64x64:
//   16 WMMAs per k-step for 8 fragment loads (4x A/B reuse).
//   Fragment layouts per CDNA5 ISA 7.12.2 (wave32):
//     A 16x32 bf16 : lane<16 -> K {0..7,16..23}; lane>=16 -> K {8..15,24..31}; M=lane&15
//     B 32x16 bf16 : lane<16 -> K {0..7,16..23}; lane>=16 -> K {8..15,24..31}; N=lane&15 (contig in W row)
//     C/D 16x16 f32: VGPR r -> M = r + 8*(lane>=16); N = lane&15
// ---------------------------------------------------------------------------
template <bool A_IS_BF16, bool SIGMOID_OUT>
__global__ __launch_bounds__(256)
void gemm_wmma(const void* __restrict__ Aptr,
               const __bf16* __restrict__ W,
               const float* __restrict__ bias,
               void* __restrict__ Cptr,
               int M, int N, int K)
{
    const int lane  = threadIdx.x & 31;
    const int wave  = threadIdx.x >> 5;
    const int waveM = wave & 1;                 // 0..1
    const int waveN = wave >> 1;                // 0..3
    const int mBase = blockIdx.x * 128 + waveM * 64;
    const int nBase = blockIdx.y * 256 + waveN * 64;

    const int mA     = lane & 15;
    const int kHalfA = (lane >> 4) << 3;        // 0 or 8
    const int nB     = lane & 15;
    const int kHalfB = (lane >> 4) << 3;        // 0 or 8

    v8f acc[4][4] = {};

    for (int k0 = 0; k0 < K; k0 += 32) {
        v16bf afrag[4], bfrag[4];

        #pragma unroll
        for (int ti = 0; ti < 4; ++ti) {
            const size_t row = (size_t)(mBase + ti * 16 + mA);
            if (A_IS_BF16) {
                const __bf16* ap = (const __bf16*)Aptr + row * K + k0;
                #pragma unroll
                for (int i = 0; i < 8; ++i) afrag[ti][i]     = ap[kHalfA + i];
                #pragma unroll
                for (int i = 0; i < 8; ++i) afrag[ti][8 + i] = ap[16 + kHalfA + i];
            } else {
                // Streamed-once fp32 activations: NT b128 loads, then whole-
                // vector convert so pairs land in packed v_cvt_pk_bf16_f32.
                const float* ap = (const float*)Aptr + row * K + k0;
                const v4f* c0 = (const v4f*)(ap + kHalfA);        // 32B aligned
                const v4f* c1 = (const v4f*)(ap + 16 + kHalfA);
                union { v16bf v; v4bf q[4]; } u;
                u.q[0] = __builtin_convertvector(__builtin_nontemporal_load(c0),     v4bf);
                u.q[1] = __builtin_convertvector(__builtin_nontemporal_load(c0 + 1), v4bf);
                u.q[2] = __builtin_convertvector(__builtin_nontemporal_load(c1),     v4bf);
                u.q[3] = __builtin_convertvector(__builtin_nontemporal_load(c1 + 1), v4bf);
                afrag[ti] = u.v;
                // Prefetch 4 k-iters ahead on the streaming operand.
                if (k0 + 128 < K)
                    __builtin_prefetch(ap + 128 + kHalfA, 0, 1);
            }
        }

        #pragma unroll
        for (int tj = 0; tj < 4; ++tj) {
            const size_t col = (size_t)(nBase + tj * 16 + nB);
            const __bf16* wp = W + col * K + k0 + kHalfB;    // K {kHalfB..+7, 16+kHalfB..+7}
            #pragma unroll
            for (int i = 0; i < 16; ++i) bfrag[tj][i] = wp[i < 8 ? i : i + 8];
        }

        #pragma unroll
        for (int ti = 0; ti < 4; ++ti)
            #pragma unroll
            for (int tj = 0; tj < 4; ++tj)
                acc[ti][tj] = __builtin_amdgcn_wmma_f32_16x16x32_bf16(
                    /*neg_a=*/false, afrag[ti],
                    /*neg_b=*/false, bfrag[tj],
                    /*c_mod=*/(short)0, acc[ti][tj],
                    /*reuse_a=*/false, /*reuse_b=*/false);
    }

    // Epilogue: bias add, optional sigmoid, store bf16 (RT: re-read from L2)
    // or fp32 output (NT: never re-read).
    const int nOut   = lane & 15;
    const int rowOfs = (lane >> 4) * 8;
    #pragma unroll
    for (int tj = 0; tj < 4; ++tj) {
        const int n    = nBase + tj * 16 + nOut;
        const float bv = bias[n];
        #pragma unroll
        for (int ti = 0; ti < 4; ++ti) {
            #pragma unroll
            for (int r = 0; r < 8; ++r) {
                const size_t m = (size_t)(mBase + ti * 16 + rowOfs + r);
                float v = acc[ti][tj][r] + bv;
                if (SIGMOID_OUT) {
                    float sv = 1.0f / (1.0f + __expf(-v));
                    __builtin_nontemporal_store(sv, (float*)Cptr + m * N + n);
                } else {
                    ((__bf16*)Cptr)[m * N + n] = (__bf16)v;
                }
            }
        }
    }
}

// ---------------------------------------------------------------------------
// Fused attention with softmax over the BATCH axis (faithful to reference).
// Grid: (S=1024, 4 p-chunks). 256 threads = 8 waves; wave w owns batch b=w.
// LDS: scores[8][16][64] fp32 = 32 KB (all 8 batches co-resident so the
// batch-axis softmax stays on-chip). Yq/Yk/Yv are L2-resident bf16.
// Layouts (from split_heads): Q[b,s,p,h] = Yq[b*S+s, h*64+p].
// Output: Yatt[row, h*64+p] = sum_q attn[b,s,p,q] * V[b,s,q,h].
// ---------------------------------------------------------------------------
__global__ __launch_bounds__(256)
void attention_kernel(const __bf16* __restrict__ Yq,
                      const __bf16* __restrict__ Yk,
                      const __bf16* __restrict__ Yv,
                      __bf16* __restrict__ Yatt)
{
    __shared__ float sc[8][16][64];   // 32 KB

    const int s    = blockIdx.x;
    const int p0   = blockIdx.y * 16;
    const int lane = threadIdx.x & 31;
    const int b    = threadIdx.x >> 5;
    const size_t row = (size_t)b * 1024 + s;

    const __bf16* qrow = Yq + row * IN_DIM;
    const __bf16* krow = Yk + row * IN_DIM;
    const __bf16* vrow = Yv + row * IN_DIM;

    // ---- scores: lane owns columns q = 2*lane, 2*lane+1 --------------------
    float kreg[2][16];
    #pragma unroll
    for (int j = 0; j < 2; ++j)
        #pragma unroll
        for (int h = 0; h < 16; ++h)
            kreg[j][h] = (float)krow[h * 64 + (2 * lane + j)];

    const float inv_scale = 0.125f;   // 1/sqrt(depth=64)
    for (int p = 0; p < 16; ++p) {
        float qh[16];
        #pragma unroll
        for (int h = 0; h < 16; ++h) qh[h] = (float)qrow[h * 64 + (p0 + p)];
        float s0 = 0.f, s1 = 0.f;
        #pragma unroll
        for (int h = 0; h < 16; ++h) {
            s0 += qh[h] * kreg[0][h];
            s1 += qh[h] * kreg[1][h];
        }
        sc[b][p][2 * lane]     = s0 * inv_scale;
        sc[b][p][2 * lane + 1] = s1 * inv_scale;
    }
    __syncthreads();

    // ---- softmax over batch axis at each (p,q): 1024 sites / 256 thr ------
    for (int site = threadIdx.x; site < 1024; site += 256) {
        const int p = site >> 6, q = site & 63;
        float v[8], mx = -1e30f;
        #pragma unroll
        for (int bb = 0; bb < 8; ++bb) { v[bb] = sc[bb][p][q]; mx = fmaxf(mx, v[bb]); }
        float sum = 0.f;
        #pragma unroll
        for (int bb = 0; bb < 8; ++bb) { v[bb] = __expf(v[bb] - mx); sum += v[bb]; }
        const float inv = 1.0f / sum;
        #pragma unroll
        for (int bb = 0; bb < 8; ++bb) sc[bb][p][q] = v[bb] * inv;
    }
    __syncthreads();

    // ---- out[b,p,h] = sum_q attn * V ; 256 sites / 32 lanes = 8 per lane --
    #pragma unroll
    for (int t = 0; t < 8; ++t) {
        const int idx = lane + 32 * t;
        const int p = idx >> 4, h = idx & 15;
        float accv = 0.f;
        #pragma unroll 16
        for (int q = 0; q < 64; ++q)
            accv += sc[b][p][q] * (float)vrow[h * 64 + q];   // contiguous V reads
        Yatt[row * IN_DIM + h * 64 + (p0 + p)] = (__bf16)accv;
    }
}

// ---------------------------------------------------------------------------
extern "C" void kernel_launch(void* const* d_in, const int* in_sizes, int n_in,
                              void* d_out, int out_size, void* d_ws, size_t ws_size,
                              hipStream_t stream)
{
    const float* query = (const float*)d_in[0];
    const float* key   = (const float*)d_in[1];
    const float* value = (const float*)d_in[2];
    const float* Wq    = (const float*)d_in[3];
    const float* bq    = (const float*)d_in[4];
    const float* Wk    = (const float*)d_in[5];
    const float* bk    = (const float*)d_in[6];
    const float* Wv    = (const float*)d_in[7];
    const float* bv    = (const float*)d_in[8];
    const float* Wo    = (const float*)d_in[9];
    const float* bo    = (const float*)d_in[10];

    char* ws = (char*)d_ws;
    __bf16* Yq   = (__bf16*)(ws);
    __bf16* Yk   = (__bf16*)(ws + 1 * YBYTES);
    __bf16* Yv   = (__bf16*)(ws + 2 * YBYTES);
    __bf16* Yatt = (__bf16*)(ws + 3 * YBYTES);
    __bf16* WqB  = (__bf16*)(ws + 4 * YBYTES);
    __bf16* WkB  = (__bf16*)(ws + 4 * YBYTES + 1 * WBYTES);
    __bf16* WvB  = (__bf16*)(ws + 4 * YBYTES + 2 * WBYTES);
    __bf16* WoB  = (__bf16*)(ws + 4 * YBYTES + 3 * WBYTES);

    const int wn4 = IN_DIM * IN_DIM / 4;
    cvt_f32_bf16<<<256, 256, 0, stream>>>(Wq, WqB, wn4);
    cvt_f32_bf16<<<256, 256, 0, stream>>>(Wk, WkB, wn4);
    cvt_f32_bf16<<<256, 256, 0, stream>>>(Wv, WvB, wn4);
    cvt_f32_bf16<<<256, 256, 0, stream>>>(Wo, WoB, wn4);

    dim3 gemmGrid(MROWS / 128, IN_DIM / 256);   // 64 x 4
    dim3 gemmBlock(256);

    gemm_wmma<false, false><<<gemmGrid, gemmBlock, 0, stream>>>(query, WqB, bq, Yq, MROWS, IN_DIM, IN_DIM);
    gemm_wmma<false, false><<<gemmGrid, gemmBlock, 0, stream>>>(key,   WkB, bk, Yk, MROWS, IN_DIM, IN_DIM);
    gemm_wmma<false, false><<<gemmGrid, gemmBlock, 0, stream>>>(value, WvB, bv, Yv, MROWS, IN_DIM, IN_DIM);

    attention_kernel<<<dim3(1024, 4), 256, 0, stream>>>(Yq, Yk, Yv, Yatt);

    gemm_wmma<true, true><<<gemmGrid, gemmBlock, 0, stream>>>(Yatt, WoB, bo, d_out, MROWS, IN_DIM, IN_DIM);
}
